// our_net_37821482009203
// MI455X (gfx1250) — 16-bit and 8-bit matrix operand order repaired and checked by reading, compiled, not run on hardware
//
#include <hip/hip_runtime.h>
#include <math.h>

typedef __attribute__((ext_vector_type(16))) _Float16 v16h;
typedef __attribute__((ext_vector_type(8)))  float    v8f;
typedef __attribute__((ext_vector_type(8)))  int      v8i;

#define BNN_EPS 1e-5f

static __device__ __forceinline__ signed char sgn8(float v) {
  return (signed char)(v > 0.f ? 1 : (v < 0.f ? -1 : 0));
}

__global__ void k_zero_f32(float* p, int n) {
  int i = blockIdx.x * blockDim.x + threadIdx.x;
  if (i < n) p[i] = 0.f;
}

__global__ void k_sign_i8(const float* __restrict__ w, signed char* __restrict__ o, int n) {
  int i = blockIdx.x * blockDim.x + threadIdx.x;
  if (i < n) o[i] = sgn8(w[i]);
}

__global__ void k_wconv_bin(const float* __restrict__ w, signed char* __restrict__ o,
                            int Cout, int Cin) {
  int K = Cin * 9;
  int total = Cout * K;
  int i = blockIdx.x * blockDim.x + threadIdx.x;
  if (i >= total) return;
  int co = i / K, k = i - co * K;
  int rs = k / Cin, ci = k - rs * Cin;
  o[i] = sgn8(w[((size_t)co * Cin + ci) * 9 + rs]);
}

__global__ void k_w1_f16(const float* __restrict__ w1, _Float16* __restrict__ o) {
  int i = blockIdx.x * blockDim.x + threadIdx.x;
  if (i < 192 * 32) {
    int r = i >> 5, k = i & 31;
    float v = (k < 27) ? w1[r * 27 + k] : 0.f;
    o[i] = (_Float16)(v > 0.f ? 1.f : (v < 0.f ? -1.f : 0.f));
  }
}

__global__ void k_fw1_bin(const float* __restrict__ fw1, signed char* __restrict__ o) {
  int i = blockIdx.x * blockDim.x + threadIdx.x;
  if (i >= 512 * 2048) return;
  int no = i >> 11, kp = i & 2047;
  int p = kp >> 9, c = kp & 511;
  o[i] = sgn8(fw1[no * 2048 + c * 4 + p]);
}

__global__ void k_fw3_pad(const float* __restrict__ fw3, const float* __restrict__ fb3,
                          signed char* __restrict__ wo, float* __restrict__ bo) {
  int i = blockIdx.x * blockDim.x + threadIdx.x;
  if (i < 16 * 512) {
    int r = i >> 9, c = i & 511;
    wo[i] = (r < 10) ? sgn8(fw3[r * 512 + c]) : (signed char)0;
    if (i < 16) bo[i] = (i < 10) ? fb3[i] : 0.f;
  }
}

__global__ void k_pool_stats(const float* __restrict__ in, float* __restrict__ pooled,
                             float* __restrict__ sum, float* __restrict__ sq,
                             int N, int C, int H, int W) {
  int OH = H >> 1, OW = W >> 1;
  int total = N * C * OH * OW;
  int i = blockIdx.x * blockDim.x + threadIdx.x;
  if (i >= total) return;
  int c = i % C; int t = i / C;
  int ow = t % OW; t /= OW;
  int oh = t % OH; int n = t / OH;
  const float* p = in + (((size_t)n * H + oh * 2) * W + (size_t)ow * 2) * C + c;
  size_t rstep = (size_t)W * C;
  float m = fmaxf(fmaxf(p[0], p[C]), fmaxf(p[rstep], p[rstep + C]));
  pooled[i] = m;
  atomicAdd(&sum[c], m);
  atomicAdd(&sq[c], m * m);
}

__global__ void k_fc_stats(const float* __restrict__ x, float* __restrict__ sum,
                           float* __restrict__ sq, int M, int Nf) {
  int i = blockIdx.x * blockDim.x + threadIdx.x;
  if (i >= M * Nf) return;
  int f = i % Nf;
  float v = x[i];
  atomicAdd(&sum[f], v);
  atomicAdd(&sq[f], v * v);
}

__global__ void k_finalize_stats(const float* __restrict__ sum, const float* __restrict__ sq,
                                 float* __restrict__ mean, float* __restrict__ inv,
                                 int C, float count) {
  int c = threadIdx.x;
  if (c < C) {
    float m = sum[c] / count;
    float v = sq[c] / count - m * m;
    mean[c] = m;
    inv[c]  = rsqrtf(v + BNN_EPS);
  }
}

__global__ void k_bn_sign(const float* __restrict__ x, const float* __restrict__ mean,
                          const float* __restrict__ inv, const float* __restrict__ g,
                          const float* __restrict__ be, signed char* __restrict__ o,
                          int C, int total) {
  int i = blockIdx.x * blockDim.x + threadIdx.x;
  if (i >= total) return;
  int c = i % C;
  float y = (x[i] - mean[c]) * inv[c] * g[c] + be[c];
  o[i] = sgn8(y);
}

__global__ void __launch_bounds__(128)
k_conv1_f16(const float* __restrict__ x, const _Float16* __restrict__ w1h,
            const float* __restrict__ bias, float* __restrict__ out,
            int N, int H, int W, int Cout) {
  __shared__ unsigned int Ah[256];
  const int HW  = H * W;
  const int tid = threadIdx.x, lane = tid & 31, wv = tid >> 5;
  const int m0  = blockIdx.x * 16;
  const int n0  = blockIdx.y * 64 + wv * 16;
  const int lh  = lane >> 4;
  const int lm  = lane & 15;

  for (int it = 0; it < 2; ++it) {
    int idx = tid * 2 + it;
    int m = idx >> 4, dw = idx & 15;
    int pos = m0 + m;
    int n = pos / HW, rem = pos - n * HW;
    int oh = rem / W, ow = rem - oh * W;
    union { _Float16 h[2]; unsigned int u; } hu;
    #pragma unroll
    for (int j = 0; j < 2; ++j) {
      int k = dw * 2 + j;
      _Float16 val = (_Float16)0.f;
      if (k < 27) {
        int ci = k / 9, rs = k - ci * 9;
        int r = rs / 3 - 1, s = rs - (rs / 3) * 3 - 1;
        int ih = oh + r, iw = ow + s;
        if (ih >= 0 && ih < H && iw >= 0 && iw < W)
          val = (_Float16)x[((size_t)n * 3 + ci) * HW + (size_t)ih * W + iw];
      }
      hu.h[j] = val;
    }
    Ah[idx] = hu.u;
  }
  __syncthreads();

  v8i ai, bi;
  #pragma unroll
  for (int vv = 0; vv < 8; ++vv) {
    int dwa = (vv < 4) ? (vv + lh * 4) : (vv + 4 + lh * 4);
    ai[vv] = (int)Ah[lm * 16 + dwa];
  }
  const unsigned int* wrow = (const unsigned int*)w1h + (size_t)(n0 + lm) * 16;
  #pragma unroll
  for (int vv = 0; vv < 8; ++vv) bi[vv] = (int)wrow[(vv < 4) ? (vv + lh * 4) : (vv + 4 + lh * 4)];

  v16h a = __builtin_bit_cast(v16h, ai);
  v16h b = __builtin_bit_cast(v16h, bi);
  v8f c = {};
  c = __builtin_amdgcn_wmma_f32_16x16x32_f16(false, a, false, b, (short)0, c, false, false);

  #pragma unroll
  for (int r = 0; r < 8; ++r) {
    int pos = m0 + r + lh * 8;
    int ch = n0 + lm;
    out[(size_t)pos * Cout + ch] = c[r] + bias[ch];
  }
}

__global__ void __launch_bounds__(128)
k_convbin_iu8(const signed char* __restrict__ in, const signed char* __restrict__ wb,
              const float* __restrict__ bias, float* __restrict__ out,
              int N, int Cin, int H, int W, int Cout) {
  __shared__ int4 As4[128];
  const int K   = Cin * 9;
  const int HW  = H * W;
  const int tid = threadIdx.x, lane = tid & 31, wv = tid >> 5;
  const int m0  = blockIdx.x * 32;
  const int n0  = blockIdx.y * 64 + wv * 16;
  const int lh  = lane >> 4;
  const int lm  = lane & 15;

  const unsigned int* wrow = (const unsigned int*)wb + (size_t)(n0 + lm) * (K >> 2);

  const int sm = tid >> 2, sq = tid & 3;
  const int spos = m0 + sm;
  const int sn = spos / HW, srem = spos - sn * HW;
  const int soh = srem / W, sow = srem - soh * W;

  v8i c0 = {}, c1 = {};
  for (int kb = 0; kb < K; kb += 64) {
    __builtin_prefetch(wrow + ((kb + 64) >> 2), 0, 0);
    int rs = kb / Cin;
    int cb = kb - rs * Cin;
    int r = rs / 3 - 1, s = rs - (rs / 3) * 3 - 1;
    int ih = soh + r, iw = sow + s;
    int4 v = make_int4(0, 0, 0, 0);
    if (ih >= 0 && ih < H && iw >= 0 && iw < W) {
      const int4* p = (const int4*)(in + ((size_t)sn * HW + (size_t)ih * W + iw) * Cin + cb);
      v = p[sq];
    }
    As4[tid] = v;
    __syncthreads();

    const unsigned int* As = (const unsigned int*)As4;
    v8i a0, a1, bv;
    #pragma unroll
    for (int vv = 0; vv < 8; ++vv) {
      int ka = (vv & 1) * 4 + ((vv >> 1) & 1) * 16 + (vv >> 2) * 32 + lh * 8;
      a0[vv] = (int)As[lm * 16 + (ka >> 2)];
      a1[vv] = (int)As[(16 + lm) * 16 + (ka >> 2)];
      int kn = (vv & 1) * 4 + ((vv >> 1) & 1) * 16 + (vv >> 2) * 32 + lh * 8;
      bv[vv] = (int)wrow[(kb + kn) >> 2];
    }
    c0 = __builtin_amdgcn_wmma_i32_16x16x64_iu8(true, a0, true, bv, c0, false, false);
    c1 = __builtin_amdgcn_wmma_i32_16x16x64_iu8(true, a1, true, bv, c1, false, false);
    __syncthreads();
  }

  #pragma unroll
  for (int r = 0; r < 8; ++r) {
    int ch = n0 + lm;
    float bs = bias[ch];
    int p0 = m0 + r + lh * 8;
    out[(size_t)p0 * Cout + ch] = (float)c0[r] + bs;
    out[(size_t)(p0 + 16) * Cout + ch] = (float)c1[r] + bs;
  }
}

__global__ void __launch_bounds__(32)
k_fcbin_iu8(const signed char* __restrict__ A, const signed char* __restrict__ B,
            const float* __restrict__ bias, float* __restrict__ out,
            int M, int K, int Nout) {
  const int lane = threadIdx.x & 31;
  const int lh = lane >> 4;
  const int lm = lane & 15;
  const int m0 = blockIdx.x * 32;
  const int n0 = blockIdx.y * 16;
  const unsigned int* arow0 = (const unsigned int*)A + (size_t)(m0 + lm) * (K >> 2);
  const unsigned int* arow1 = (const unsigned int*)A + (size_t)(m0 + 16 + lm) * (K >> 2);
  const unsigned int* brow  = (const unsigned int*)B + (size_t)(n0 + lm) * (K >> 2);
  v8i c0 = {}, c1 = {};
  for (int kb = 0; kb < K; kb += 64) {
    __builtin_prefetch(brow + ((kb + 64) >> 2), 0, 0);
    v8i a0, a1, bv;
    #pragma unroll
    for (int vv = 0; vv < 8; ++vv) {
      int ka = (vv & 1) * 4 + ((vv >> 1) & 1) * 16 + (vv >> 2) * 32 + lh * 8;
      a0[vv] = (int)arow0[(kb + ka) >> 2];
      a1[vv] = (int)arow1[(kb + ka) >> 2];
      int kn = (vv & 1) * 4 + ((vv >> 1) & 1) * 16 + (vv >> 2) * 32 + lh * 8;
      bv[vv] = (int)brow[(kb + kn) >> 2];
    }
    c0 = __builtin_amdgcn_wmma_i32_16x16x64_iu8(true, a0, true, bv, c0, false, false);
    c1 = __builtin_amdgcn_wmma_i32_16x16x64_iu8(true, a1, true, bv, c1, false, false);
  }
  #pragma unroll
  for (int r = 0; r < 8; ++r) {
    int n = n0 + lm;
    float bs = bias[n];
    int mr = m0 + r + lh * 8;
    out[(size_t)mr * Nout + n] = (float)c0[r] + bs;
    out[(size_t)(mr + 16) * Nout + n] = (float)c1[r] + bs;
  }
}

__global__ void __launch_bounds__(256)
k_final(const float* __restrict__ in, float* __restrict__ out) {
  __shared__ float s[10], q[10];
  int t = threadIdx.x;
  if (t < 10) { s[t] = 0.f; q[t] = 0.f; }
  __syncthreads();
  float v[10];
  #pragma unroll
  for (int j = 0; j < 10; ++j) {
    float x = in[t * 16 + j];
    v[j] = x;
    atomicAdd(&s[j], x);
    atomicAdd(&q[j], x * x);
  }
  __syncthreads();
  float xh[10], mx = -1e30f;
  #pragma unroll
  for (int j = 0; j < 10; ++j) {
    float m  = s[j] * (1.f / 256.f);
    float vr = q[j] * (1.f / 256.f) - m * m;
    xh[j] = (v[j] - m) * rsqrtf(vr + BNN_EPS);
    mx = fmaxf(mx, xh[j]);
  }
  float se = 0.f;
  #pragma unroll
  for (int j = 0; j < 10; ++j) se += __expf(xh[j] - mx);
  float lse = mx + __logf(se);
  #pragma unroll
  for (int j = 0; j < 10; ++j) out[t * 10 + j] = xh[j] - lse;
}

extern "C" void kernel_launch(void* const* d_in, const int* in_sizes, int n_in,
                              void* d_out, int out_size, void* d_ws, size_t ws_size,
                              hipStream_t stream) {
  const float* x   = (const float*)d_in[0];
  const float* w1  = (const float*)d_in[1];  const float* b1  = (const float*)d_in[2];
  const float* g1  = (const float*)d_in[3];  const float* be1 = (const float*)d_in[4];
  const float* w2  = (const float*)d_in[5];  const float* b2  = (const float*)d_in[6];
  const float* g2  = (const float*)d_in[7];  const float* be2 = (const float*)d_in[8];
  const float* w3  = (const float*)d_in[9];  const float* b3  = (const float*)d_in[10];
  const float* g3  = (const float*)d_in[11]; const float* be3 = (const float*)d_in[12];
  const float* w4  = (const float*)d_in[13]; const float* b4  = (const float*)d_in[14];
  const float* g4  = (const float*)d_in[15]; const float* be4 = (const float*)d_in[16];
  const float* fw1 = (const float*)d_in[17]; const float* fb1 = (const float*)d_in[18];
  const float* g5  = (const float*)d_in[19]; const float* be5 = (const float*)d_in[20];
  const float* fw2 = (const float*)d_in[21]; const float* fb2 = (const float*)d_in[22];
  const float* g6  = (const float*)d_in[23]; const float* be6 = (const float*)d_in[24];
  const float* fw3 = (const float*)d_in[25]; const float* fb3 = (const float*)d_in[26];
  float* outp = (float*)d_out;

  const int Nb = 256, C1 = 192, C2 = 384, C3 = 768, C4 = 512;

  char* ws = (char*)d_ws;
  size_t off = 0;
  auto take = [&](size_t bytes) -> char* {
    char* p = ws + off;
    off = (off + bytes + 255) & ~(size_t)255;
    return p;
  };
  float*       F    = (float*)take((size_t)Nb * C1 * 32 * 32 * 4);
  float*       POOL = (float*)take((size_t)Nb * C1 * 16 * 16 * 4);
  signed char* BIN  = (signed char*)take((size_t)Nb * C1 * 16 * 16);
  _Float16*    W1H  = (_Float16*)take((size_t)C1 * 32 * 2);
  signed char* W2B  = (signed char*)take((size_t)C2 * C1 * 9);
  signed char* W3B  = (signed char*)take((size_t)C3 * C2 * 9);
  signed char* W4B  = (signed char*)take((size_t)C4 * C3 * 9);
  signed char* FW1B = (signed char*)take((size_t)512 * 2048);
  signed char* FW2B = (signed char*)take((size_t)512 * 512);
  signed char* FW3B = (signed char*)take((size_t)16 * 512);
  float*       FB3P = (float*)take(16 * 4);
  float*       SUM  = (float*)take(768 * 4);
  float*       SQ   = (float*)take(768 * 4);
  float*       MEAN = (float*)take(768 * 4);
  float*       INV  = (float*)take(768 * 4);
  float*       FC1  = (float*)take((size_t)256 * 512 * 4);
  signed char* FCB1 = (signed char*)take((size_t)256 * 512);
  float*       FC2  = (float*)take((size_t)256 * 512 * 4);
  signed char* FCB2 = (signed char*)take((size_t)256 * 512);
  float*       FC3  = (float*)take((size_t)256 * 16 * 4);
  (void)ws_size; (void)n_in; (void)in_sizes; (void)out_size;

  auto blocks = [](int n) { return (n + 255) / 256; };

  k_w1_f16<<<blocks(C1 * 32), 256, 0, stream>>>(w1, W1H);
  k_wconv_bin<<<blocks(C2 * C1 * 9), 256, 0, stream>>>(w2, W2B, C2, C1);
  k_wconv_bin<<<blocks(C3 * C2 * 9), 256, 0, stream>>>(w3, W3B, C3, C2);
  k_wconv_bin<<<blocks(C4 * C3 * 9), 256, 0, stream>>>(w4, W4B, C4, C3);
  k_fw1_bin<<<blocks(512 * 2048), 256, 0, stream>>>(fw1, FW1B);
  k_sign_i8<<<blocks(512 * 512), 256, 0, stream>>>(fw2, FW2B, 512 * 512);
  k_fw3_pad<<<blocks(16 * 512), 256, 0, stream>>>(fw3, fb3, FW3B, FB3P);

  k_conv1_f16<<<dim3((Nb * 32 * 32) / 16, C1 / 64), 128, 0, stream>>>(x, W1H, b1, F, Nb, 32, 32, C1);
  k_zero_f32<<<blocks(1536), 256, 0, stream>>>(SUM, 1536);
  k_pool_stats<<<blocks(Nb * C1 * 16 * 16), 256, 0, stream>>>(F, POOL, SUM, SQ, Nb, C1, 32, 32);
  k_finalize_stats<<<1, C1, 0, stream>>>(SUM, SQ, MEAN, INV, C1, (float)(Nb * 16 * 16));
  k_bn_sign<<<blocks(Nb * C1 * 16 * 16), 256, 0, stream>>>(POOL, MEAN, INV, g1, be1, BIN, C1, Nb * C1 * 16 * 16);

  k_convbin_iu8<<<dim3((Nb * 16 * 16) / 32, C2 / 64), 128, 0, stream>>>(BIN, W2B, b2, F, Nb, C1, 16, 16, C2);
  k_zero_f32<<<blocks(1536), 256, 0, stream>>>(SUM, 1536);
  k_pool_stats<<<blocks(Nb * C2 * 8 * 8), 256, 0, stream>>>(F, POOL, SUM, SQ, Nb, C2, 16, 16);
  k_finalize_stats<<<1, C2, 0, stream>>>(SUM, SQ, MEAN, INV, C2, (float)(Nb * 8 * 8));
  k_bn_sign<<<blocks(Nb * C2 * 8 * 8), 256, 0, stream>>>(POOL, MEAN, INV, g2, be2, BIN, C2, Nb * C2 * 8 * 8);

  k_convbin_iu8<<<dim3((Nb * 8 * 8) / 32, C3 / 64), 128, 0, stream>>>(BIN, W3B, b3, F, Nb, C2, 8, 8, C3);
  k_zero_f32<<<blocks(1536), 256, 0, stream>>>(SUM, 1536);
  k_pool_stats<<<blocks(Nb * C3 * 4 * 4), 256, 0, stream>>>(F, POOL, SUM, SQ, Nb, C3, 8, 8);
  k_finalize_stats<<<1, C3, 0, stream>>>(SUM, SQ, MEAN, INV, C3, (float)(Nb * 4 * 4));
  k_bn_sign<<<blocks(Nb * C3 * 4 * 4), 256, 0, stream>>>(POOL, MEAN, INV, g3, be3, BIN, C3, Nb * C3 * 4 * 4);

  k_convbin_iu8<<<dim3((Nb * 4 * 4) / 32, C4 / 64), 128, 0, stream>>>(BIN, W4B, b4, F, Nb, C3, 4, 4, C4);
  k_zero_f32<<<blocks(1536), 256, 0, stream>>>(SUM, 1536);
  k_pool_stats<<<blocks(Nb * C4 * 2 * 2), 256, 0, stream>>>(F, POOL, SUM, SQ, Nb, C4, 4, 4);
  k_finalize_stats<<<1, C4, 0, stream>>>(SUM, SQ, MEAN, INV, C4, (float)(Nb * 2 * 2));
  k_bn_sign<<<blocks(Nb * C4 * 2 * 2), 256, 0, stream>>>(POOL, MEAN, INV, g4, be4, BIN, C4, Nb * C4 * 2 * 2);

  k_fcbin_iu8<<<dim3(256 / 32, 512 / 16), 32, 0, stream>>>(BIN, FW1B, fb1, FC1, 256, 2048, 512);
  k_zero_f32<<<blocks(1536), 256, 0, stream>>>(SUM, 1536);
  k_fc_stats<<<blocks(256 * 512), 256, 0, stream>>>(FC1, SUM, SQ, 256, 512);
  k_finalize_stats<<<1, 512, 0, stream>>>(SUM, SQ, MEAN, INV, 512, 256.f);
  k_bn_sign<<<blocks(256 * 512), 256, 0, stream>>>(FC1, MEAN, INV, g5, be5, FCB1, 512, 256 * 512);

  k_fcbin_iu8<<<dim3(256 / 32, 512 / 16), 32, 0, stream>>>(FCB1, FW2B, fb2, FC2, 256, 512, 512);
  k_zero_f32<<<blocks(1536), 256, 0, stream>>>(SUM, 1536);
  k_fc_stats<<<blocks(256 * 512), 256, 0, stream>>>(FC2, SUM, SQ, 256, 512);
  k_finalize_stats<<<1, 512, 0, stream>>>(SUM, SQ, MEAN, INV, 512, 256.f);
  k_bn_sign<<<blocks(256 * 512), 256, 0, stream>>>(FC2, MEAN, INV, g6, be6, FCB2, 512, 256 * 512);

  k_fcbin_iu8<<<dim3(256 / 32, 1), 32, 0, stream>>>(FCB2, FW3B, FB3P, FC3, 256, 512, 16);
  k_final<<<1, 256, 0, stream>>>(FC3, outp);
}
